// ASAP_58033598104017
// MI455X (gfx1250) — matrix operand order repaired and checked by reading, not compiled, not run on hardware
//
#include <hip/hip_runtime.h>
#include <math.h>

typedef __attribute__((ext_vector_type(16))) __bf16 v16bf;
typedef __attribute__((ext_vector_type(8)))  __bf16 v8bf;
typedef __attribute__((ext_vector_type(8)))  float  v8f;

#define GN 50000
#define GE 1200000
#define GBATCH 8
#define INV_STD_H 0.99999500003749972f   // 1/sqrt(1+1e-5)

// ---------------- fragment loaders (contiguous-K 16-bit A/B layouts) ----------------

__device__ __forceinline__ v16bf frag_combine(v8bf lo, v8bf hi) {
  v16bf o;
#pragma unroll
  for (int i = 0; i < 8; ++i) { o[i] = lo[i]; o[i + 8] = hi[i]; }
  return o;
}

// A 16x32 bf16: lane holds row m=lane&15, K = 32*ks + e + 16*half (16 contiguous)
__device__ __forceinline__ v16bf load_fragA(const __bf16* row, int ks, int half) {
  v8bf lo = *(const v8bf*)(row + 32 * ks + 16 * half);
  v8bf hi = *(const v8bf*)(row + 32 * ks + 16 * half + 8);
  return frag_combine(lo, hi);
}

// B 32x16 bf16 from transposed weights wt[N][KW]: lane holds col N=lane&15,
// K = 32*ks + e + 16*half  (16 contiguous bf16)
__device__ __forceinline__ v16bf load_fragB(const __bf16* wrow, int ks, int half) {
  const __bf16* p = wrow + 32 * ks + 16 * half;
  v8bf lo = *(const v8bf*)(p);
  v8bf hi = *(const v8bf*)(p + 8);
  return frag_combine(lo, hi);
}

__device__ __forceinline__ void atomicMaxF32(float* addr, float v) {
  if (v >= 0.f) atomicMax((int*)addr, __float_as_int(v));
  else          atomicMin((unsigned int*)addr, __float_as_uint(v));
}

// ---------------- one MLP layer: [16,32*KS] x [32*KS,64] via WMMA ----------

template<int KS, bool LAST>
__device__ __forceinline__ void mlp_layer(
    const __bf16* inRow, const __bf16* wt,
    const float* bias, const float* scale, const float* beta,
    __bf16* outBuf, float* agg, const int* dstIdx, int lane)
{
  const int half = lane >> 4, lm = lane & 15;
  v16bf a[KS];
#pragma unroll
  for (int ks = 0; ks < KS; ++ks) a[ks] = load_fragA(inRow, ks, half);
#pragma unroll
  for (int nt = 0; nt < 4; ++nt) {
    const int n = nt * 16 + lm;
    v8f c = {};
#pragma unroll
    for (int ks = 0; ks < KS; ++ks) {
      v16bf b = load_fragB(wt + n * (32 * KS), ks, half);
      c = __builtin_amdgcn_wmma_f32_16x16x32_bf16(false, a[ks], false, b,
                                                  (short)0, c, false, false);
    }
    const float bi = bias[n], sc = scale[n], bt = beta[n];
#pragma unroll
    for (int r = 0; r < 8; ++r) {
      const int m = r + 8 * half;                 // D layout: M = r + 8*half
      float v = c[r] + bi;
      v = fmaxf(v, 0.f) * sc + bt;                // ReLU then eval-mode BN
      if (LAST) {
        int d = dstIdx[m];
        if (d >= 0) atomicMaxF32(agg + (size_t)d * 64 + n, v);
      } else {
        outBuf[m * 64 + n] = (__bf16)v;
      }
    }
  }
}

// ---------------- EdgeConv: gather -> 3-layer MLP -> scatter-max ----------

template<int KPAD0, bool CONV2>
__global__ __launch_bounds__(128) void edgeconv_kernel(
    const int* __restrict__ src, const int* __restrict__ dst,
    const float* __restrict__ pos, const float* __restrict__ xin,
    const __bf16* __restrict__ h1bf,
    const float* __restrict__ W0g, const float* __restrict__ b0g,
    const float* __restrict__ g0g, const float* __restrict__ be0g,
    const float* __restrict__ W1g, const float* __restrict__ b1g,
    const float* __restrict__ g1g, const float* __restrict__ be1g,
    const float* __restrict__ W2g, const float* __restrict__ b2g,
    const float* __restrict__ g2g, const float* __restrict__ be2g,
    float* __restrict__ agg, int E, int numTiles)
{
  constexpr int WAVES = 4;
  constexpr int IN0 = CONV2 ? 70 : 9;
  constexpr int KS0 = KPAD0 / 32;

  __shared__ __align__(16) __bf16 sW0[64 * KPAD0];   // transposed [n][k]
  __shared__ __align__(16) __bf16 sW1[64 * 64];
  __shared__ __align__(16) __bf16 sW2[64 * 64];
  __shared__ float sBias[3][64], sScale[3][64], sBeta[3][64];
  __shared__ __align__(16) __bf16 sIn[WAVES][16 * KPAD0];
  __shared__ __align__(16) __bf16 sMidA[WAVES][16 * 64];
  __shared__ __align__(16) __bf16 sMidB[WAVES][16 * 64];
  __shared__ int sDst[WAVES][16];

  const int tid = threadIdx.x;
  for (int i = tid; i < 64 * KPAD0; i += 128) {
    int n = i / KPAD0, k = i % KPAD0;
    sW0[i] = (__bf16)((k < IN0) ? W0g[k * 64 + n] : 0.f);
  }
  for (int i = tid; i < 64 * 64; i += 128) {
    int n = i >> 6, k = i & 63;
    sW1[i] = (__bf16)W1g[k * 64 + n];
    sW2[i] = (__bf16)W2g[k * 64 + n];
  }
  for (int i = tid; i < 64; i += 128) {
    sBias[0][i] = b0g[i]; sScale[0][i] = INV_STD_H * g0g[i]; sBeta[0][i] = be0g[i];
    sBias[1][i] = b1g[i]; sScale[1][i] = INV_STD_H * g1g[i]; sBeta[1][i] = be1g[i];
    sBias[2][i] = b2g[i]; sScale[2][i] = INV_STD_H * g2g[i]; sBeta[2][i] = be2g[i];
  }
  __syncthreads();

  const int wave = tid >> 5, lane = tid & 31, lm = lane & 15;
  const int gw = blockIdx.x * WAVES + wave;
  const int step = gridDim.x * WAVES;

  for (int t = gw; t < numTiles; t += step) {
    const int base = t * 16;
    if (lane < 16) {
      const int e = base + lane;
      const bool valid = (e < E);
      int s = 0, d = 0;
      if (valid) { s = src[e]; d = dst[e]; }
      sDst[wave][lane] = valid ? d : -1;
      __bf16* row = &sIn[wave][lane * KPAD0];
      float p0 = 0, p1 = 0, p2 = 0, q0 = 0, q1 = 0, q2 = 0;
      if (valid) {
        p0 = pos[3 * d]; p1 = pos[3 * d + 1]; p2 = pos[3 * d + 2];
        q0 = pos[3 * s] - p0; q1 = pos[3 * s + 1] - p1; q2 = pos[3 * s + 2] - p2;
      }
      row[0] = (__bf16)p0; row[1] = (__bf16)p1; row[2] = (__bf16)p2;
      row[3] = (__bf16)q0; row[4] = (__bf16)q1; row[5] = (__bf16)q2;
      if (!CONV2) {
        float x0 = 0, x1 = 0, x2 = 0;
        if (valid) { x0 = xin[3 * d]; x1 = xin[3 * d + 1]; x2 = xin[3 * d + 2]; }
        row[6] = (__bf16)x0; row[7] = (__bf16)x1; row[8] = (__bf16)x2;
        for (int k = 9; k < KPAD0; ++k) row[k] = (__bf16)0.f;
      } else {
        const __bf16* hp = h1bf + (size_t)d * 64;
        for (int k = 0; k < 64; ++k) row[6 + k] = valid ? hp[k] : (__bf16)0.f;
        for (int k = IN0; k < KPAD0; ++k) row[k] = (__bf16)0.f;
      }
      __builtin_prefetch((const void*)(src + base + 16), 0, 0);
    }
    mlp_layer<KS0, false>(&sIn[wave][lm * KPAD0], sW0, sBias[0], sScale[0], sBeta[0],
                          &sMidA[wave][0], nullptr, nullptr, lane);
    mlp_layer<2, false>(&sMidA[wave][lm * 64], sW1, sBias[1], sScale[1], sBeta[1],
                        &sMidB[wave][0], nullptr, nullptr, lane);
    mlp_layer<2, true>(&sMidB[wave][lm * 64], sW2, sBias[2], sScale[2], sBeta[2],
                       nullptr, agg, sDst[wave], lane);
  }
}

// ---------------- init / finalize / head ----------------

__global__ void init_kernel(float* agg, int nAgg, float* zbuf, int nZero) {
  int i = blockIdx.x * blockDim.x + threadIdx.x;
  if (i < nAgg) ((unsigned*)agg)[i] = 0xFF800000u;           // -inf
  else if (i - nAgg < nZero) zbuf[i - nAgg] = 0.f;
}

template<bool FIRST>
__global__ void finalize_kernel(const float* __restrict__ agg,
                                __bf16* __restrict__ h1bf,
                                float* __restrict__ pool, float* __restrict__ cnt,
                                const int* __restrict__ batch, int N) {
  int idx = blockIdx.x * blockDim.x + threadIdx.x;
  if (idx >= N * 64) return;
  int n = idx >> 6, f = idx & 63;
  float v = fmaxf(agg[idx], 0.f);       // -inf (empty segment) -> 0, plus ReLU
  int b = batch[n];
  if (FIRST) h1bf[idx] = (__bf16)v;
  atomicAdd(&pool[b * 64 + f], v);
  if (FIRST && f == 0) atomicAdd(&cnt[b], 1.0f);
}

__global__ __launch_bounds__(128) void head_kernel(
    const float* __restrict__ pool1, const float* __restrict__ pool2,
    const float* __restrict__ cnt,
    const float* __restrict__ W1, const float* __restrict__ b1,
    const float* __restrict__ W2, const float* __restrict__ b2,
    float* __restrict__ out) {
  __shared__ float sP[GBATCH][128];
  __shared__ float sH[GBATCH][64];
  __shared__ float sL[GBATCH][10];
  int t = threadIdx.x;
  for (int i = t; i < GBATCH * 64; i += 128) {
    int g = i >> 6, f = i & 63;
    float c = fmaxf(cnt[g], 1.f);
    sP[g][f] = pool1[i] / c;
    sP[g][64 + f] = pool2[i] / c;
  }
  __syncthreads();
  for (int i = t; i < GBATCH * 64; i += 128) {
    int g = i >> 6, o = i & 63;
    float v = b1[o];
    for (int k = 0; k < 128; ++k) v += sP[g][k] * W1[k * 64 + o];
    sH[g][o] = fmaxf(v, 0.f);
  }
  __syncthreads();
  for (int i = t; i < GBATCH * 10; i += 128) {
    int g = i / 10, c = i % 10;
    float v = b2[c];
    for (int k = 0; k < 64; ++k) v += sH[g][k] * W2[k * 10 + c];
    sL[g][c] = v;
  }
  __syncthreads();
  if (t < GBATCH) {
    float m = -1e30f;
    for (int c = 0; c < 10; ++c) m = fmaxf(m, sL[t][c]);
    float s = 0.f;
    for (int c = 0; c < 10; ++c) s += __expf(sL[t][c] - m);
    float lse = m + __logf(s);
    for (int c = 0; c < 10; ++c) out[t * 10 + c] = sL[t][c] - lse;
  }
}

// ---------------- host-side orchestration ----------------

extern "C" void kernel_launch(void* const* d_in, const int* in_sizes, int n_in,
                              void* d_out, int out_size, void* d_ws, size_t ws_size,
                              hipStream_t stream) {
  (void)in_sizes; (void)n_in; (void)out_size; (void)ws_size;
  const int N = GN, E = GE, B = GBATCH;

  const float* x     = (const float*)d_in[0];
  const float* pos   = (const float*)d_in[1];
  const int*   ei    = (const int*)d_in[2];
  const int*   batch = (const int*)d_in[3];
  const int* src = ei;
  const int* dst = ei + E;

  // conv1 blocks: d_in[4..15], conv2 blocks: d_in[16..27], lin1: 28/29, lin2: 30/31
  const float* c1[12]; for (int i = 0; i < 12; ++i) c1[i] = (const float*)d_in[4 + i];
  const float* c2[12]; for (int i = 0; i < 12; ++i) c2[i] = (const float*)d_in[16 + i];
  const float* lin1W = (const float*)d_in[28];
  const float* lin1b = (const float*)d_in[29];
  const float* lin2W = (const float*)d_in[30];
  const float* lin2b = (const float*)d_in[31];

  // workspace: agg[N*64] f32 | h1bf[N*64] bf16 | pool1[512] | pool2[512] | cnt[8]
  float*  agg   = (float*)d_ws;
  __bf16* h1bf  = (__bf16*)((char*)d_ws + (size_t)N * 64 * 4);
  float*  pool1 = (float*)((char*)h1bf + (size_t)N * 64 * 2);
  float*  pool2 = pool1 + B * 64;
  float*  cnt   = pool2 + B * 64;

  const int nAgg = N * 64;
  const int nZero = B * 64 * 2 + B;
  const int numTiles = (E + 15) / 16;

  init_kernel<<<(nAgg + nZero + 255) / 256, 256, 0, stream>>>(agg, nAgg, pool1, nZero);

  edgeconv_kernel<32, false><<<4096, 128, 0, stream>>>(
      src, dst, pos, x, (const __bf16*)nullptr,
      c1[0], c1[1], c1[2], c1[3], c1[4], c1[5], c1[6], c1[7], c1[8], c1[9], c1[10], c1[11],
      agg, E, numTiles);

  finalize_kernel<true><<<(nAgg + 255) / 256, 256, 0, stream>>>(
      agg, h1bf, pool1, cnt, batch, N);

  init_kernel<<<(nAgg + 255) / 256, 256, 0, stream>>>(agg, nAgg, nullptr, 0);

  edgeconv_kernel<96, true><<<4096, 128, 0, stream>>>(
      src, dst, pos, (const float*)nullptr, h1bf,
      c2[0], c2[1], c2[2], c2[3], c2[4], c2[5], c2[6], c2[7], c2[8], c2[9], c2[10], c2[11],
      agg, E, numTiles);

  finalize_kernel<false><<<(nAgg + 255) / 256, 256, 0, stream>>>(
      agg, (__bf16*)nullptr, pool2, (float*)nullptr, batch, N);

  head_kernel<<<1, 128, 0, stream>>>(pool1, pool2, cnt, lin1W, lin1b, lin2W, lin2b,
                                     (float*)d_out);
}
